// Model_38019050505046
// MI455X (gfx1250) — matrix operand order repaired and checked by reading, not compiled, not run on hardware
//
#include <hip/hip_runtime.h>
#include <hip/hip_bf16.h>

typedef __attribute__((ext_vector_type(16))) _Float16 v16h;
typedef __attribute__((ext_vector_type(8)))  _Float16 v8h;
typedef __attribute__((ext_vector_type(8)))  float    v8f;

#define T_LEN   4096
#define FLAT_D  512
#define HD_D    256
#define FEAT_D  20
#define MAXW_D  20
#define NCAND_D 30000
#define HID_D   512
#define KSEL    1638
#define C_ANT   40
#define MENT_D  1300   // 2*512 + 20 + 256
#define MENTP   1344   // padded to multiple of 64
#define KC_D    (KSEL * C_ANT)   // 65520

// ---------------------------------------------------------------------------
// WMMA GEMM: C = act(A[MxK] @ B[KxN] (+bias)), f16 in, f32 acc.
// 128x64 tile / 128-thread block (4 waves); each wave computes 32x64 via
// 8 accumulators. K must be a multiple of 32. transB=1: B stored NxK (A@B^T).
// LDS layouts are fragment-contiguous so fragment reads are ds_load_b128.
// ---------------------------------------------------------------------------
__global__ __launch_bounds__(128)
void k_gemm(const _Float16* __restrict__ A, int lda,
            const _Float16* __restrict__ B, int ldb, int transB,
            int M, int N, int Kd,
            const float* __restrict__ bias, int relu,
            _Float16* __restrict__ Ch, int ldch,
            float* __restrict__ Cf, int ldcf)
{
    __shared__ _Float16 sA[128 * 40];  // [row][k] rows of 32 halves, stride 40
    __shared__ _Float16 sBt[64 * 40];  // [col][k] cols of 32 halves, stride 40

    const int tid  = threadIdx.x;
    const int lane = tid & 31;
    const int wave = tid >> 5;
    const int bm = blockIdx.y * 128;
    const int bn = blockIdx.x * 64;

    v8f acc[2][4];
    for (int t = 0; t < 2; ++t)
        for (int ns = 0; ns < 4; ++ns)
            for (int e = 0; e < 8; ++e) acc[t][ns][e] = 0.0f;

    for (int k0 = 0; k0 < Kd; k0 += 32) {
        __syncthreads();
        // --- stage A tile: 128 rows x 32 halves, 16B vector loads ---
        for (int ch = tid; ch < 512; ch += 128) {
            int r  = ch >> 2;
            int c8 = (ch & 3) * 8;
            int gr = bm + r;
            v8h v;
            if (gr < M) {
                v = *(const v8h*)(A + (size_t)gr * lda + k0 + c8);
            } else {
#pragma unroll
                for (int j = 0; j < 8; ++j) v[j] = (_Float16)0.0f;
            }
            *(v8h*)&sA[r * 40 + c8] = v;
        }
        // --- stage B tile transposed into [col][k] ---
        if (!transB) {
            // B is KxN: read rows (contiguous in n), scatter-transpose to LDS
            for (int ch = tid; ch < 256; ch += 128) {
                int r  = ch >> 3;          // k within tile (0..31)
                int c8 = (ch & 7) * 8;     // col chunk
                v8h v;
                if (bn + c8 + 7 < N) {
                    v = *(const v8h*)(B + (size_t)(k0 + r) * ldb + bn + c8);
                } else {
#pragma unroll
                    for (int j = 0; j < 8; ++j) {
                        int gc = bn + c8 + j;
                        v[j] = (gc < N) ? B[(size_t)(k0 + r) * ldb + gc]
                                        : (_Float16)0.0f;
                    }
                }
#pragma unroll
                for (int j = 0; j < 8; ++j) sBt[(c8 + j) * 40 + r] = v[j];
            }
        } else {
            // B is NxK: columns of B^T are contiguous -> vector load + store
            for (int ch = tid; ch < 256; ch += 128) {
                int c  = ch >> 2;          // col (0..63)
                int r8 = (ch & 3) * 8;     // k chunk
                int gc = bn + c;
                v8h v;
                if (gc < N) {
                    v = *(const v8h*)(B + (size_t)gc * ldb + k0 + r8);
                } else {
#pragma unroll
                    for (int j = 0; j < 8; ++j) v[j] = (_Float16)0.0f;
                }
                *(v8h*)&sBt[c * 40 + r8] = v;
            }
        }
        __syncthreads();

        // --- fragments (ISA 7.12.2 16-bit layouts), contiguous b128 reads ---
        const int am    = lane & 15;
        const int khalf = lane >> 4;
        v16h af[2];
#pragma unroll
        for (int t = 0; t < 2; ++t) {
            int arow = wave * 32 + t * 16 + am;
            v8h a0 = *(const v8h*)&sA[arow * 40 + khalf * 8];
            v8h a1 = *(const v8h*)&sA[arow * 40 + 16 + khalf * 8];
            af[t] = __builtin_shufflevector(a0, a1, 0, 1, 2, 3, 4, 5, 6, 7,
                                            8, 9, 10, 11, 12, 13, 14, 15);
        }

#pragma unroll
        for (int ns = 0; ns < 4; ++ns) {
            int bcol = ns * 16 + am;
            v8h b0 = *(const v8h*)&sBt[bcol * 40 + khalf * 8];
            v8h b1 = *(const v8h*)&sBt[bcol * 40 + 16 + khalf * 8];
            v16h bf = __builtin_shufflevector(b0, b1, 0, 1, 2, 3, 4, 5, 6, 7,
                                              8, 9, 10, 11, 12, 13, 14, 15);
            acc[0][ns] = __builtin_amdgcn_wmma_f32_16x16x32_f16(
                false, af[0], false, bf, (short)0, acc[0][ns], false, false);
            acc[1][ns] = __builtin_amdgcn_wmma_f32_16x16x32_f16(
                false, af[1], false, bf, (short)0, acc[1][ns], false, false);
        }
    }

    // C/D layout: VGPR e -> row e (+8 for lanes 16-31), col = lane&15
#pragma unroll
    for (int t = 0; t < 2; ++t) {
#pragma unroll
        for (int ns = 0; ns < 4; ++ns) {
#pragma unroll
            for (int e = 0; e < 8; ++e) {
                int row = bm + wave * 32 + t * 16 + ((lane >> 4) * 8) + e;
                int col = bn + ns * 16 + (lane & 15);
                if (row < M && col < N) {
                    float v = acc[t][ns][e];
                    if (bias) v += bias[col];
                    if (relu && v < 0.0f) v = 0.0f;
                    if (Ch) Ch[(size_t)row * ldch + col] = (_Float16)v;
                    if (Cf) Cf[(size_t)row * ldcf + col] = v;
                }
            }
        }
    }
}

// ---------------------------------------------------------------------------
// Small helper kernels
// ---------------------------------------------------------------------------
__global__ void k_convert_pad(const float* __restrict__ src, int srows, int scols,
                              _Float16* __restrict__ dst, int drows, int dcols)
{
    int i = blockIdx.x * blockDim.x + threadIdx.x;
    if (i >= drows * dcols) return;
    int r = i / dcols, c = i % dcols;
    float v = (r < srows && c < scols) ? src[(size_t)r * scols + c] : 0.0f;
    dst[i] = (_Float16)v;
}

__global__ void k_pad_f32(const float* __restrict__ src, int n,
                          float* __restrict__ dst, int np)
{
    int i = blockIdx.x * blockDim.x + threadIdx.x;
    if (i >= np) return;
    dst[i] = (i < n) ? src[i] : 0.0f;
}

__global__ void k_head_scores(const float* __restrict__ flat,
                              const float* __restrict__ w,
                              const float* __restrict__ b,
                              float* __restrict__ hs)
{
    int t = blockIdx.x * blockDim.x + threadIdx.x;
    if (t >= T_LEN) return;
    float s = b[0];
    for (int d = 0; d < FLAT_D; ++d) s += flat[(size_t)t * FLAT_D + d] * w[d];
    hs[t] = s;
}

// one wave (32 lanes) per candidate: build [start|end|width|head] embedding
__global__ __launch_bounds__(128)
void k_cand_build(const float* __restrict__ flat, const float* __restrict__ hdoc,
                  const float* __restrict__ hs, const float* __restrict__ widthE,
                  const int* __restrict__ cs, const int* __restrict__ ce,
                  float* __restrict__ candF, _Float16* __restrict__ candH)
{
    int n = blockIdx.x * 4 + (threadIdx.x >> 5);
    if (n >= NCAND_D) return;
    int lane = threadIdx.x & 31;
    int s = cs[n], e = ce[n];
    int w = e - s + 1;
    size_t bF = (size_t)n * MENT_D, bH = (size_t)n * MENTP;

    for (int d = lane; d < FLAT_D; d += 32) {
        float v1 = flat[(size_t)s * FLAT_D + d];
        float v2 = flat[(size_t)e * FLAT_D + d];
        candF[bF + d] = v1;            candH[bH + d] = (_Float16)v1;
        candF[bF + FLAT_D + d] = v2;   candH[bH + FLAT_D + d] = (_Float16)v2;
    }
    for (int d = lane; d < FEAT_D; d += 32) {
        float v = widthE[(w - 1) * FEAT_D + d];
        candF[bF + 2 * FLAT_D + d] = v; candH[bH + 2 * FLAT_D + d] = (_Float16)v;
    }
    float p[MAXW_D];
    float mx = -3.0e38f;
    for (int j = 0; j < MAXW_D; ++j) {
        int idx = s + j; if (idx > T_LEN - 1) idx = T_LEN - 1;
        float g = hs[idx] + ((j < w) ? 0.0f : -__builtin_inff());
        p[j] = g; if (g > mx) mx = g;
    }
    float Z = 0.0f;
    for (int j = 0; j < MAXW_D; ++j) { float ev = expf(p[j] - mx); p[j] = ev; Z += ev; }
    float inv = 1.0f / Z;
    for (int d = lane; d < HD_D; d += 32) {
        float a = 0.0f;
        for (int j = 0; j < MAXW_D; ++j) {
            int idx = s + j; if (idx > T_LEN - 1) idx = T_LEN - 1;
            a += p[j] * hdoc[(size_t)idx * HD_D + d];
        }
        a *= inv;
        candF[bF + 2 * FLAT_D + FEAT_D + d] = a;
        candH[bH + 2 * FLAT_D + FEAT_D + d] = (_Float16)a;
    }
    for (int d = MENT_D + lane; d < MENTP; d += 32) candH[bH + d] = (_Float16)0.0f;
}

// final scorer layer (N=1) as a wave-per-row dot product, optional addend
__global__ void k_dot(const _Float16* __restrict__ H, int ld, int Kd,
                      const float* __restrict__ w, const float* __restrict__ b1,
                      const float* __restrict__ addend, float* __restrict__ out, int M)
{
    int wave = threadIdx.x >> 5, lane = threadIdx.x & 31;
    int r = blockIdx.x * 4 + wave;
    if (r >= M) return;
    float s = 0.0f;
    for (int j = lane; j < Kd; j += 32) s += (float)H[(size_t)r * ld + j] * w[j];
    for (int off = 16; off; off >>= 1) s += __shfl_xor(s, off);
    if (lane == 0) out[r] = s + b1[0] + (addend ? addend[r] : 0.0f);
}

__global__ void k_make_keys(const float* __restrict__ sc, unsigned* __restrict__ keys)
{
    int i = blockIdx.x * blockDim.x + threadIdx.x;
    if (i >= NCAND_D) return;
    unsigned u = __float_as_uint(sc[i]);
    keys[i] = (u & 0x80000000u) ? ~u : (u | 0x80000000u);
}

// single-block 4-bit radix select: finds Kth-largest key; ints={thr,numTies,ctr}
__global__ __launch_bounds__(512)
void k_radix_select(const unsigned* __restrict__ keys, int n, int Ksel,
                    unsigned* __restrict__ ints)
{
    __shared__ int cnt[16];
    __shared__ unsigned sprefix;
    __shared__ int sremaining;
    int tid = threadIdx.x;
    if (tid == 0) { sprefix = 0; sremaining = Ksel; }
    __syncthreads();
    for (int shift = 28; shift >= 0; shift -= 4) {
        if (tid < 16) cnt[tid] = 0;
        __syncthreads();
        unsigned pref = sprefix;
        for (int i = tid; i < n; i += 512) {
            unsigned k = keys[i];
            bool match = (shift == 28) || ((k >> (shift + 4)) == pref);
            if (match) atomicAdd(&cnt[(k >> shift) & 15], 1);
        }
        __syncthreads();
        if (tid == 0) {
            int acc = 0, rem = sremaining, sel = 0;
            for (int d = 15; d >= 0; --d) {
                if (acc + cnt[d] >= rem) { sel = d; break; }
                acc += cnt[d];
            }
            sremaining = rem - acc;
            sprefix = (sprefix << 4) | (unsigned)sel;
        }
        __syncthreads();
    }
    if (tid == 0) { ints[0] = sprefix; ints[1] = (unsigned)sremaining; ints[2] = 0; }
}

__global__ void k_collect(const unsigned* __restrict__ keys, int n,
                          unsigned* __restrict__ ints, int* __restrict__ sel)
{
    int i = blockIdx.x * blockDim.x + threadIdx.x;
    if (i >= n) return;
    unsigned thr = ints[0];
    int numTies = (int)ints[1];
    unsigned k = keys[i];
    if (k > thr) {
        int p = (int)atomicAdd(&ints[2], 1u);
        sel[p] = i;
    } else if (k == thr) {
        int r = 0;
        for (int j = 0; j < i; ++j) r += (keys[j] == thr);
        if (r < numTies) {
            int p = (int)atomicAdd(&ints[2], 1u);
            sel[p] = i;
        }
    }
}

// single-block bitonic sort (2048 slots) of selected mentions by ment_pos
__global__ __launch_bounds__(1024)
void k_sort(const int* __restrict__ sel, int Ksel,
            const int* __restrict__ cs, const int* __restrict__ ce,
            int ncand, int* __restrict__ outIdx)
{
    __shared__ unsigned long long arr[2048];
    int tid = threadIdx.x;
    long long lastEnd = (long long)ce[ncand - 1];
    for (int i = tid; i < 2048; i += 1024) {
        if (i < Ksel) {
            int id = sel[i];
            unsigned long long pos =
                (unsigned long long)((long long)cs[id] * lastEnd + ce[id]);
            arr[i] = (pos << 32) | (unsigned)id;
        } else arr[i] = ~0ull;
    }
    __syncthreads();
    for (int k2 = 2; k2 <= 2048; k2 <<= 1)
        for (int j = k2 >> 1; j > 0; j >>= 1) {
            for (int t = tid; t < 2048; t += 1024) {
                int ixj = t ^ j;
                if (ixj > t) {
                    bool up = ((t & k2) == 0);
                    unsigned long long a = arr[t], b = arr[ixj];
                    if ((a > b) == up) { arr[t] = b; arr[ixj] = a; }
                }
            }
            __syncthreads();
        }
    for (int i = tid; i < Ksel; i += 1024)
        outIdx[i] = (int)(arr[i] & 0xffffffffu);
}

__global__ __launch_bounds__(128)
void k_gather(const int* __restrict__ topIdx, const float* __restrict__ candF,
              const float* __restrict__ candScores, const int* __restrict__ cs,
              const int* __restrict__ spkIds,
              float* __restrict__ mentF, _Float16* __restrict__ mentH,
              float* __restrict__ mentScores, int* __restrict__ mentSpk)
{
    int k = blockIdx.x, tid = threadIdx.x;
    int id = topIdx[k];
    for (int d = tid; d < MENT_D; d += 128) {
        float v = candF[(size_t)id * MENT_D + d];
        mentF[(size_t)k * MENT_D + d] = v;
        mentH[(size_t)k * MENTP + d] = (_Float16)v;
    }
    for (int d = MENT_D + tid; d < MENTP; d += 128)
        mentH[(size_t)k * MENTP + d] = (_Float16)0.0f;
    if (tid == 0) {
        mentScores[k] = candScores[id];
        mentSpk[k] = spkIds[cs[id]];
    }
}

__global__ void k_fast_epi(float* __restrict__ fastM, int ldf,
                           const float* __restrict__ ms)
{
    int i = blockIdx.x * blockDim.x + threadIdx.x;
    if (i >= KSEL * KSEL) return;
    int row = i / KSEL, col = i % KSEL;
    float v = fastM[(size_t)row * ldf + col] + ms[row] + ms[col];
    if (!(row - col > 0)) v = -__builtin_inff();
    fastM[(size_t)row * ldf + col] = v;
}

// per-row top-C (sorted desc, smallest-index tie-break like lax.top_k)
__global__ __launch_bounds__(256)
void k_topc(const float* __restrict__ fastM, int ldf,
            float* __restrict__ topFast, int* __restrict__ antes)
{
    __shared__ float vals[1664];
    __shared__ unsigned char selm[1664];
    __shared__ float rv[256];
    __shared__ int ri[256];
    int row = blockIdx.x, tid = threadIdx.x;
    for (int j = tid; j < KSEL; j += 256) {
        vals[j] = fastM[(size_t)row * ldf + j];
        selm[j] = 0;
    }
    __syncthreads();
    for (int c = 0; c < C_ANT; ++c) {
        float best = -__builtin_inff();
        int bidx = 0x7fffffff;
        for (int j = tid; j < KSEL; j += 256) {
            if (!selm[j]) {
                float v = vals[j];
                if (v > best || (v == best && j < bidx)) { best = v; bidx = j; }
            }
        }
        rv[tid] = best; ri[tid] = bidx;
        __syncthreads();
        for (int s = 128; s; s >>= 1) {
            if (tid < s) {
                if (rv[tid + s] > rv[tid] ||
                    (rv[tid + s] == rv[tid] && ri[tid + s] < ri[tid])) {
                    rv[tid] = rv[tid + s]; ri[tid] = ri[tid + s];
                }
            }
            __syncthreads();
        }
        if (tid == 0) {
            int b = (ri[0] == 0x7fffffff) ? 0 : ri[0];
            topFast[row * C_ANT + c] = rv[0];
            antes[row * C_ANT + c] = b;
            selm[b] = 1;
        }
        __syncthreads();
    }
}

// precompute feature->HID tables: spk(2x512), genre(+b1)(512), dist(9x512)
__global__ __launch_bounds__(512)
void k_tables(const float* __restrict__ ssw1, const float* __restrict__ ssb1,
              const float* __restrict__ spkE, const float* __restrict__ genE,
              const float* __restrict__ distE, const int* __restrict__ genre_id,
              float* __restrict__ spkTab, float* __restrict__ genreVec,
              float* __restrict__ distTab)
{
    int n = threadIdx.x;
    const int base = 3 * MENT_D;   // 3900
    float s0 = 0.0f, s1 = 0.0f;
    for (int f = 0; f < FEAT_D; ++f) {
        s0 += spkE[f] * ssw1[(size_t)(base + f) * HID_D + n];
        s1 += spkE[FEAT_D + f] * ssw1[(size_t)(base + f) * HID_D + n];
    }
    spkTab[n] = s0; spkTab[HID_D + n] = s1;
    int gid = genre_id[0];
    float g = ssb1[n];
    for (int f = 0; f < FEAT_D; ++f)
        g += genE[gid * FEAT_D + f] * ssw1[(size_t)(base + FEAT_D + f) * HID_D + n];
    genreVec[n] = g;
    for (int d = 0; d < 9; ++d) {
        float t = 0.0f;
        for (int f = 0; f < FEAT_D; ++f)
            t += distE[d * FEAT_D + f] * ssw1[(size_t)(base + 2 * FEAT_D + f) * HID_D + n];
        distTab[d * HID_D + n] = t;
    }
}

__global__ __launch_bounds__(256)
void k_prod_build(const float* __restrict__ mentF, const int* __restrict__ antes,
                  _Float16* __restrict__ prodH)
{
    int r = blockIdx.x;               // k*C + c
    int k = r / C_ANT;
    int a = antes[r];
    for (int d = threadIdx.x; d < MENTP; d += 256) {
        float v = (d < MENT_D)
            ? mentF[(size_t)k * MENT_D + d] * mentF[(size_t)a * MENT_D + d]
            : 0.0f;
        prodH[(size_t)r * MENTP + d] = (_Float16)v;
    }
}

__device__ __forceinline__ int dist_bin(int d)
{
    if (d < 0) d = 0; if (d > 64) d = 64;
    if (d <= 4) return d;
    if (d <= 8) return 5;
    if (d <= 16) return 6;
    if (d <= 32) return 7;
    return 8;
}

__global__ __launch_bounds__(512)
void k_h1_assemble(const float* __restrict__ prodOut, const float* __restrict__ meProj,
                   const float* __restrict__ amProj, const int* __restrict__ antes,
                   const int* __restrict__ mentSpk, const float* __restrict__ spkTab,
                   const float* __restrict__ genreVec, const float* __restrict__ distTab,
                   _Float16* __restrict__ h1H)
{
    int r = blockIdx.x, n = threadIdx.x;
    int k = r / C_ANT;
    int a = antes[r];
    int same = (mentSpk[k] == mentSpk[a]) ? 1 : 0;
    int bin = dist_bin(k - a);
    float v = prodOut[(size_t)r * HID_D + n]
            + meProj[(size_t)k * HID_D + n]
            + amProj[(size_t)a * HID_D + n]
            + spkTab[same * HID_D + n]
            + genreVec[n]
            + distTab[bin * HID_D + n];
    h1H[(size_t)r * HID_D + n] = (_Float16)(v > 0.0f ? v : 0.0f);
}

__global__ __launch_bounds__(128)
void k_attend(const float* __restrict__ mentF, const int* __restrict__ antes,
              const float* __restrict__ anteScores,
              float* __restrict__ attF, _Float16* __restrict__ attH)
{
    int k = blockIdx.x, tid = threadIdx.x;
    float w[C_ANT + 1];
    w[0] = 0.0f;
    float mx = 0.0f;
    for (int c = 0; c < C_ANT; ++c) {
        float v = anteScores[k * C_ANT + c];
        w[1 + c] = v; if (v > mx) mx = v;
    }
    float Z = 0.0f;
    for (int c = 0; c <= C_ANT; ++c) { float e = expf(w[c] - mx); w[c] = e; Z += e; }
    float inv = 1.0f / Z;
    for (int d = tid; d < MENT_D; d += 128) {
        float a = w[0] * mentF[(size_t)k * MENT_D + d];
        for (int c = 0; c < C_ANT; ++c)
            a += w[1 + c] * mentF[(size_t)antes[k * C_ANT + c] * MENT_D + d];
        a *= inv;
        attF[(size_t)k * MENT_D + d] = a;
        attH[(size_t)k * MENTP + d] = (_Float16)a;
    }
    for (int d = MENT_D + tid; d < MENTP; d += 128)
        attH[(size_t)k * MENTP + d] = (_Float16)0.0f;
}

__global__ void k_gate_update(const float* __restrict__ g1, const float* __restrict__ g2,
                              const float* __restrict__ attF,
                              float* __restrict__ mentF, _Float16* __restrict__ mentH)
{
    int i = blockIdx.x * blockDim.x + threadIdx.x;
    if (i >= KSEL * MENT_D) return;
    int k = i / MENT_D, d = i % MENT_D;
    float f = 1.0f / (1.0f + expf(-(g1[i] + g2[i])));
    float nm = f * attF[i] + (1.0f - f) * mentF[i];
    mentF[i] = nm;
    mentH[(size_t)k * MENTP + d] = (_Float16)nm;
}

__global__ void k_final_out(const float* __restrict__ anteScores, float* __restrict__ out)
{
    int i = blockIdx.x * blockDim.x + threadIdx.x;
    if (i >= KSEL * (C_ANT + 1)) return;
    int k = i / (C_ANT + 1), c = i % (C_ANT + 1);
    out[i] = (c == 0) ? 0.0f : anteScores[k * C_ANT + (c - 1)];
}

// ---------------------------------------------------------------------------
// Orchestration
// ---------------------------------------------------------------------------
extern "C" void kernel_launch(void* const* d_in, const int* in_sizes, int n_in,
                              void* d_out, int out_size, void* d_ws, size_t ws_size,
                              hipStream_t stream)
{
    const float* flat   = (const float*)d_in[0];
    const float* hdoc   = (const float*)d_in[1];
    const float* wsh    = (const float*)d_in[2];
    const float* bsh    = (const float*)d_in[3];
    const float* widthE = (const float*)d_in[4];
    const float* ms_w1  = (const float*)d_in[5];
    const float* ms_b1  = (const float*)d_in[6];
    const float* ms_w2  = (const float*)d_in[7];
    const float* ms_b2  = (const float*)d_in[8];
    const float* ms_w3  = (const float*)d_in[9];
    const float* ms_b3  = (const float*)d_in[10];
    const float* fast_w = (const float*)d_in[11];
    const float* fast_b = (const float*)d_in[12];
    const float* ss_w1  = (const float*)d_in[13];
    const float* ss_b1  = (const float*)d_in[14];
    const float* ss_w2  = (const float*)d_in[15];
    const float* ss_b2  = (const float*)d_in[16];
    const float* ss_w3  = (const float*)d_in[17];
    const float* ss_b3  = (const float*)d_in[18];
    const float* gate_w = (const float*)d_in[19];
    const float* gate_b = (const float*)d_in[20];
    const float* spkE   = (const float*)d_in[21];
    const float* genE   = (const float*)d_in[22];
    const float* distE  = (const float*)d_in[23];
    const int*   cs     = (const int*)d_in[24];
    const int*   ce     = (const int*)d_in[25];
    const int*   spkIds = (const int*)d_in[26];
    const int*   genreId= (const int*)d_in[27];
    float* out = (float*)d_out;

    // ---- workspace carve-up ----
    char* p = (char*)d_ws;
    auto alloc = [&](size_t bytes) -> void* {
        void* r = (void*)p;
        p += (bytes + 255) & ~(size_t)255;
        return r;
    };
    float*     headScores = (float*)alloc((size_t)T_LEN * 4);
    float*     candF      = (float*)alloc((size_t)NCAND_D * MENT_D * 4);
    _Float16*  candH      = (_Float16*)alloc((size_t)NCAND_D * MENTP * 2);
    _Float16*  h1H        = (_Float16*)alloc((size_t)KC_D * HID_D * 2);
    _Float16*  h2H        = (_Float16*)alloc((size_t)KC_D * HID_D * 2);
    float*     candScores = (float*)alloc((size_t)NCAND_D * 4);
    unsigned*  keys       = (unsigned*)alloc((size_t)NCAND_D * 4);
    unsigned*  ints       = (unsigned*)alloc(64);
    int*       selIdx     = (int*)alloc(2048 * 4);
    int*       topIdx     = (int*)alloc(2048 * 4);
    float*     mentF      = (float*)alloc((size_t)KSEL * MENT_D * 4);
    _Float16*  mentH      = (_Float16*)alloc((size_t)KSEL * MENTP * 2);
    float*     mentScores = (float*)alloc((size_t)KSEL * 4);
    int*       mentSpk    = (int*)alloc((size_t)KSEL * 4);
    _Float16*  srcH       = (_Float16*)alloc((size_t)KSEL * MENTP * 2);
    const int  LDF = 1664;
    float*     fastM      = (float*)alloc((size_t)KSEL * LDF * 4);
    float*     topFast    = (float*)alloc((size_t)KC_D * 4);
    int*       antes      = (int*)alloc((size_t)KC_D * 4);
    float*     anteScores = (float*)alloc((size_t)KC_D * 4);
    float*     meProj     = (float*)alloc((size_t)KSEL * HID_D * 4);
    float*     amProj     = (float*)alloc((size_t)KSEL * HID_D * 4);
    _Float16*  prodH      = (_Float16*)alloc((size_t)KC_D * MENTP * 2);
    float*     prodOut    = (float*)alloc((size_t)KC_D * HID_D * 4);
    float*     attF       = (float*)alloc((size_t)KSEL * MENT_D * 4);
    _Float16*  attH       = (_Float16*)alloc((size_t)KSEL * MENTP * 2);
    float*     g1         = (float*)alloc((size_t)KSEL * MENT_D * 4);
    float*     g2         = (float*)alloc((size_t)KSEL * MENT_D * 4);
    float*     spkTab     = (float*)alloc(2 * HID_D * 4);
    float*     genreVec   = (float*)alloc(HID_D * 4);
    float*     distTab    = (float*)alloc(9 * HID_D * 4);
    float*     fastBpad   = (float*)alloc(MENTP * 4);
    _Float16*  w1H        = (_Float16*)alloc((size_t)MENTP * HID_D * 2);
    _Float16*  w2H        = (_Float16*)alloc((size_t)HID_D * HID_D * 2);
    _Float16*  fastWH     = (_Float16*)alloc((size_t)MENTP * MENTP * 2);
    _Float16*  ssWmeH     = (_Float16*)alloc((size_t)MENTP * HID_D * 2);
    _Float16*  ssWamH     = (_Float16*)alloc((size_t)MENTP * HID_D * 2);
    _Float16*  ssWprodH   = (_Float16*)alloc((size_t)MENTP * HID_D * 2);
    _Float16*  ssW2H      = (_Float16*)alloc((size_t)HID_D * HID_D * 2);
    _Float16*  gateTopH   = (_Float16*)alloc((size_t)MENTP * MENT_D * 2);
    _Float16*  gateBotH   = (_Float16*)alloc((size_t)MENTP * MENT_D * 2);

    auto cdiv = [](int a, int b) { return (a + b - 1) / b; };
    auto conv = [&](const float* s, int sr, int sc, _Float16* d, int dr, int dc) {
        int tot = dr * dc;
        k_convert_pad<<<cdiv(tot, 256), 256, 0, stream>>>(s, sr, sc, d, dr, dc);
    };

    // ---- weight conversion (f32 -> f16, K padded to 1344 where needed) ----
    conv(ms_w1, MENT_D, HID_D, w1H, MENTP, HID_D);
    conv(ms_w2, HID_D, HID_D, w2H, HID_D, HID_D);
    conv(fast_w, MENT_D, MENT_D, fastWH, MENTP, MENTP);
    conv(ss_w1,                    MENT_D, HID_D, ssWmeH,   MENTP, HID_D);
    conv(ss_w1 + (size_t)MENT_D * HID_D,     MENT_D, HID_D, ssWamH,   MENTP, HID_D);
    conv(ss_w1 + (size_t)2 * MENT_D * HID_D, MENT_D, HID_D, ssWprodH, MENTP, HID_D);
    conv(ss_w2, HID_D, HID_D, ssW2H, HID_D, HID_D);
    conv(gate_w,                         MENT_D, MENT_D, gateTopH, MENTP, MENT_D);
    conv(gate_w + (size_t)MENT_D * MENT_D, MENT_D, MENT_D, gateBotH, MENTP, MENT_D);
    k_pad_f32<<<cdiv(MENTP, 256), 256, 0, stream>>>(fast_b, MENT_D, fastBpad, MENTP);

    // ---- candidate embeddings + mention scorer ----
    k_head_scores<<<cdiv(T_LEN, 256), 256, 0, stream>>>(flat, wsh, bsh, headScores);
    k_cand_build<<<cdiv(NCAND_D, 4), 128, 0, stream>>>(flat, hdoc, headScores, widthE,
                                                       cs, ce, candF, candH);
    k_gemm<<<dim3(cdiv(HID_D, 64), cdiv(NCAND_D, 128)), 128, 0, stream>>>(
        candH, MENTP, w1H, HID_D, 0, NCAND_D, HID_D, MENTP, ms_b1, 1,
        h1H, HID_D, nullptr, 0);
    k_gemm<<<dim3(cdiv(HID_D, 64), cdiv(NCAND_D, 128)), 128, 0, stream>>>(
        h1H, HID_D, w2H, HID_D, 0, NCAND_D, HID_D, HID_D, ms_b2, 1,
        h2H, HID_D, nullptr, 0);
    k_dot<<<cdiv(NCAND_D, 4), 128, 0, stream>>>(h2H, HID_D, HID_D, ms_w3, ms_b3,
                                                nullptr, candScores, NCAND_D);

    // ---- top-K mentions + sort by ment_pos ----
    k_make_keys<<<cdiv(NCAND_D, 256), 256, 0, stream>>>(candScores, keys);
    k_radix_select<<<1, 512, 0, stream>>>(keys, NCAND_D, KSEL, ints);
    k_collect<<<cdiv(NCAND_D, 256), 256, 0, stream>>>(keys, NCAND_D, ints, selIdx);
    k_sort<<<1, 1024, 0, stream>>>(selIdx, KSEL, cs, ce, NCAND_D, topIdx);
    k_gather<<<KSEL, 128, 0, stream>>>(topIdx, candF, candScores, cs, spkIds,
                                       mentF, mentH, mentScores, mentSpk);

    // ---- fast antecedent scores + per-row top-C ----
    k_gemm<<<dim3(cdiv(MENTP, 64), cdiv(KSEL, 128)), 128, 0, stream>>>(
        mentH, MENTP, fastWH, MENTP, 0, KSEL, MENTP, MENTP, fastBpad, 0,
        srcH, MENTP, nullptr, 0);
    k_gemm<<<dim3(cdiv(KSEL, 64), cdiv(KSEL, 128)), 128, 0, stream>>>(
        srcH, MENTP, mentH, MENTP, 1, KSEL, KSEL, MENTP, nullptr, 0,
        nullptr, 0, fastM, LDF);
    k_fast_epi<<<cdiv(KSEL * KSEL, 256), 256, 0, stream>>>(fastM, LDF, mentScores);
    k_topc<<<KSEL, 256, 0, stream>>>(fastM, LDF, topFast, antes);

    // ---- slow scorer feature tables ----
    k_tables<<<1, 512, 0, stream>>>(ss_w1, ss_b1, spkE, genE, distE, genreId,
                                    spkTab, genreVec, distTab);

    // ---- DEPTH refinement iterations ----
    const int DEPTH = 2;
    for (int dIt = 0; dIt < DEPTH; ++dIt) {
        k_gemm<<<dim3(cdiv(HID_D, 64), cdiv(KSEL, 128)), 128, 0, stream>>>(
            mentH, MENTP, ssWmeH, HID_D, 0, KSEL, HID_D, MENTP, nullptr, 0,
            nullptr, 0, meProj, HID_D);
        k_gemm<<<dim3(cdiv(HID_D, 64), cdiv(KSEL, 128)), 128, 0, stream>>>(
            mentH, MENTP, ssWamH, HID_D, 0, KSEL, HID_D, MENTP, nullptr, 0,
            nullptr, 0, amProj, HID_D);
        k_prod_build<<<KC_D, 256, 0, stream>>>(mentF, antes, prodH);
        k_gemm<<<dim3(cdiv(HID_D, 64), cdiv(KC_D, 128)), 128, 0, stream>>>(
            prodH, MENTP, ssWprodH, HID_D, 0, KC_D, HID_D, MENTP, nullptr, 0,
            nullptr, 0, prodOut, HID_D);
        k_h1_assemble<<<KC_D, 512, 0, stream>>>(prodOut, meProj, amProj, antes,
                                                mentSpk, spkTab, genreVec, distTab, h1H);
        k_gemm<<<dim3(cdiv(HID_D, 64), cdiv(KC_D, 128)), 128, 0, stream>>>(
            h1H, HID_D, ssW2H, HID_D, 0, KC_D, HID_D, HID_D, ss_b2, 1,
            h2H, HID_D, nullptr, 0);
        k_dot<<<cdiv(KC_D, 4), 128, 0, stream>>>(h2H, HID_D, HID_D, ss_w3, ss_b3,
                                                 topFast, anteScores, KC_D);
        if (dIt < DEPTH - 1) {
            k_attend<<<KSEL, 128, 0, stream>>>(mentF, antes, anteScores, attF, attH);
            k_gemm<<<dim3(cdiv(MENT_D, 64), cdiv(KSEL, 128)), 128, 0, stream>>>(
                mentH, MENTP, gateTopH, MENT_D, 0, KSEL, MENT_D, MENTP, gate_b, 0,
                nullptr, 0, g1, MENT_D);
            k_gemm<<<dim3(cdiv(MENT_D, 64), cdiv(KSEL, 128)), 128, 0, stream>>>(
                attH, MENTP, gateBotH, MENT_D, 0, KSEL, MENT_D, MENTP, nullptr, 0,
                nullptr, 0, g2, MENT_D);
            k_gate_update<<<cdiv(KSEL * MENT_D, 256), 256, 0, stream>>>(
                g1, g2, attF, mentF, mentH);
        }
    }

    k_final_out<<<cdiv(KSEL * (C_ANT + 1), 256), 256, 0, stream>>>(anteScores, out);
}
